// LoraQuantizedLinear_11227044511974
// MI455X (gfx1250) — compile-verified, matrix operand order checked by reading, not run on hardware
//
#include <hip/hip_runtime.h>
#include <stdint.h>

// ---------------- problem constants ----------------
#define TOKENS   64
#define IN_DIM   4096
#define OUT_DIM  13696
#define LORA_R   32

#define NBLK     64                  // output rows per workgroup -> 13696/64 = 214 blocks
#define KC       256                 // K chunk of x staged in LDS
#define LDS_STR  (KC + 8)            // +8 halves padding -> conflict-free ds_load_b128
#define NCHUNK   (IN_DIM / KC)       // 16 (even -> clean 2-phase unroll)

#define AS1 __attribute__((address_space(1)))
#define AS3 __attribute__((address_space(3)))

typedef _Float16 v16h __attribute__((ext_vector_type(16)));
typedef float    v8f  __attribute__((ext_vector_type(8)));
typedef _Float16 h2   __attribute__((ext_vector_type(2)));
typedef int      i32x4 __attribute__((ext_vector_type(4)));

union Frag {            // one WMMA 16-bit operand: 8 VGPRs / 16 halves per lane
    v16h     v;
    uint32_t u[8];
    uint4    q[2];
};

// ---- CDNA5 async global->LDS copy (ASYNCcnt) with portable fallback ----
__device__ __forceinline__ void async_cp_b128(const void* g, void* l) {
#if __has_builtin(__builtin_amdgcn_global_load_async_to_lds_b128)
    // probed signature: (int4 addrspace(1)*, int4 addrspace(3)*, imm offset, imm cpol)
    __builtin_amdgcn_global_load_async_to_lds_b128(
        (AS1 i32x4*)g, (AS3 i32x4*)l, 0, 0);
#else
    uint32_t loff = (uint32_t)(uintptr_t)(AS3 void*)l;
    asm volatile("global_load_async_to_lds_b128 %0, %1, off"
                 :: "v"(loff), "v"(g) : "memory");
#endif
}

template <int N>
__device__ __forceinline__ void async_wait() {
#if __has_builtin(__builtin_amdgcn_s_wait_asynccnt)
    __builtin_amdgcn_s_wait_asynccnt(N);
#else
    asm volatile("s_wait_asynccnt %0" :: "i"(N) : "memory");
#endif
}

// Dequant 8 packed signed int4 (one dword) -> 4 packed-f16 pairs.
// nibble s in [-8,7]: u = s+8 via xor of nibble sign bit; f16(0x6400|u) = 1024+u
// value = (1024+u)*scale - 1032*scale  ->  one v_pk_fma_f16 per pair.
__device__ __forceinline__ void dequant_dword(uint32_t p, h2 s2, h2 b2, uint32_t* dst) {
    uint32_t t  = p ^ 0x88888888u;
    uint32_t hn = (t >> 4) & 0x0F0F0F0Fu;     // even-K nibbles (high nibble first)
    uint32_t ln =  t       & 0x0F0F0F0Fu;     // odd-K nibbles
#pragma unroll
    for (int r = 0; r < 4; ++r) {
        uint32_t sel = 0x0C000C04u | ((uint32_t)r << 16) | (uint32_t)r;
        uint32_t raw = 0x64006400u | __builtin_amdgcn_perm(hn, ln, sel);
        h2 q = __builtin_bit_cast(h2, raw);
        h2 d = q * s2 + b2;
        dst[r] = __builtin_bit_cast(uint32_t, d);
    }
}

__device__ __forceinline__ v8f wmma_f16(const Frag& a, const Frag& b, v8f c) {
    return __builtin_amdgcn_wmma_f32_16x16x32_f16(false, a.v, false, b.v,
                                                  (short)0, c, false, false);
}

// ---------------- stage 1: tmp[64,32] = x @ lora_a^T (WMMA) ----------------
__global__ __launch_bounds__(256)
void lora_xa_kernel(const _Float16* __restrict__ x,
                    const _Float16* __restrict__ lora_a,
                    _Float16* __restrict__ tmp) {
    const int tid  = threadIdx.x;
    const int lane = tid & 31;
    const int lr   = lane & 15;
    const int hi   = lane >> 4;
    const int wave = tid >> 5;            // 8 waves: 4 M-tiles x 2 N-tiles
    const int mt   = (wave & 3) * 16;
    const int nc   = (wave >> 2) * 16;

    const _Float16* xrow = x      + (size_t)(mt + lr) * IN_DIM;
    const _Float16* arow = lora_a + (size_t)(nc + lr) * IN_DIM;

    v8f acc = {};
    for (int k = 0; k < IN_DIM; k += 32) {
        Frag a, b;
        a.q[0] = *(const uint4*)(xrow + k + 8 * hi);
        a.q[1] = *(const uint4*)(xrow + k + 16 + 8 * hi);
        b.q[0] = *(const uint4*)(arow + k + 8 * hi);
        b.q[1] = *(const uint4*)(arow + k + 16 + 8 * hi);
        acc = wmma_f16(a, b, acc);
    }
#pragma unroll
    for (int v = 0; v < 8; ++v)
        tmp[(size_t)(mt + v + 8 * hi) * LORA_R + nc + lr] = (_Float16)acc[v];
}

// ---------------- stage 2 helpers ----------------

// Issue this chunk's async stage of x[:, kc:kc+KC] into an LDS buffer (8 b128/thread).
__device__ __forceinline__ void stage_chunk(const _Float16* __restrict__ x,
                                            _Float16* __restrict__ buf,
                                            int kc, int tid) {
#pragma unroll
    for (int i = 0; i < (TOKENS * KC / 8) / 256; ++i) {   // 8 iterations
        int f   = tid + i * 256;
        int row = f >> 5;                                 // KC/8 = 32 uint4 per row
        int c   = f & 31;
        async_cp_b128(&x[(size_t)row * IN_DIM + kc + c * 8],
                      &buf[row * LDS_STR + c * 8]);
    }
}

// Load a whole chunk of this lane's int4 weight row into registers (non-temporal).
__device__ __forceinline__ void load_w(uint32_t (&pw)[16], const int8_t* wrow,
                                       int kc, int hi) {
    const int kb = kc >> 1;
#pragma unroll
    for (int i = 0; i < 8; ++i) {
        pw[2 * i]     = __builtin_nontemporal_load((const uint32_t*)(wrow + kb + 16 * i + 4 * hi));
        pw[2 * i + 1] = __builtin_nontemporal_load((const uint32_t*)(wrow + kb + 16 * i + 8 + 4 * hi));
    }
}

// Dequant + 2 M-tiles of WMMA per k-step over one staged chunk.
__device__ __forceinline__ void compute_chunk(const _Float16* __restrict__ As,
                                              const uint32_t (&pw)[16],
                                              h2 s2, h2 b2, int mh, int lr, int hi,
                                              v8f acc[2]) {
#pragma unroll
    for (int ksi = 0; ksi < KC / 32; ++ksi) {
        Frag b;
        dequant_dword(pw[2 * ksi],     s2, b2, &b.u[0]);
        dequant_dword(pw[2 * ksi + 1], s2, b2, &b.u[4]);
#pragma unroll
        for (int t = 0; t < 2; ++t) {
            Frag a;
            const _Float16* ap = &As[(mh + t * 16 + lr) * LDS_STR + ksi * 32 + 8 * hi];
            a.q[0] = *(const uint4*)ap;
            a.q[1] = *(const uint4*)(ap + 16);
            acc[t] = wmma_f16(a, b, acc[t]);
        }
    }
}

// ---------------- stage 2: int4-dequant GEMM + LoRA epilogue ----------------
__global__ __launch_bounds__(256)
void lora_qlinear_kernel(const _Float16* __restrict__ x,
                         const int8_t*   __restrict__ w,       // [OUT_DIM][IN_DIM/2] int4
                         const _Float16* __restrict__ wscale,  // [OUT_DIM]
                         const _Float16* __restrict__ lora_b,  // [OUT_DIM][LORA_R]
                         const _Float16* __restrict__ tmp,     // [TOKENS][LORA_R]
                         _Float16* __restrict__ out) {
    __shared__ _Float16 As[2][TOKENS * LDS_STR];   // 2 x 33.8 KB double buffer

    const int tid  = threadIdx.x;
    const int lane = tid & 31;
    const int wave = tid >> 5;
    const int lr   = lane & 15;
    const int hi   = lane >> 4;
    const int nt   = wave & 3;          // N-tile (4 x 16 = 64 rows per block)
    const int mh   = (wave >> 2) * 32;  // M half: 2 M-tiles per wave

    const int n = blockIdx.x * NBLK + nt * 16 + lr;   // this lane's output row
    const int8_t* wrow = w + (size_t)n * (IN_DIM / 2);

    const _Float16 s  = wscale[n];
    h2 s2; s2[0] = s; s2[1] = s;
    const _Float16 bb = (_Float16)(-1032.0f * (float)s);
    h2 b2; b2[0] = bb; b2[1] = bb;

    v8f acc[2] = {};

    // software pipeline: async LDS stage (ASYNCcnt) + register weight prefetch (LOADcnt)
    uint32_t pwA[16], pwB[16];
    stage_chunk(x, As[0], 0, tid);
    load_w(pwA, wrow, 0, hi);

    for (int c = 0; c < NCHUNK; c += 2) {
        // -------- phase 0: compute chunk c (buf0), prefetch chunk c+1 (buf1) --------
        stage_chunk(x, As[1], (c + 1) * KC, tid);   // c+1 <= 15 always
        async_wait<8>();                            // older batch (buf0) complete
        __syncthreads();                            // all waves' buf0 writes visible
        load_w(pwB, wrow, (c + 1) * KC, hi);
        compute_chunk(As[0], pwA, s2, b2, mh, lr, hi, acc);
        __syncthreads();                            // done reading buf0 before re-stage

        // -------- phase 1: compute chunk c+1 (buf1), prefetch chunk c+2 (buf0) ------
        if (c + 2 < NCHUNK) {
            stage_chunk(x, As[0], (c + 2) * KC, tid);
            async_wait<8>();
        } else {
            async_wait<0>();
        }
        __syncthreads();
        if (c + 2 < NCHUNK) load_w(pwA, wrow, (c + 2) * KC, hi);
        compute_chunk(As[1], pwB, s2, b2, mh, lr, hi, acc);
        __syncthreads();
    }

    // ---- LoRA epilogue: acc += tmp(64x32) @ lora_b^T(32x16), alpha/r = 1.0 ----
    Frag lb;
    const _Float16* lbrow = lora_b + (size_t)n * LORA_R;
    lb.q[0] = *(const uint4*)(lbrow + 8 * hi);
    lb.q[1] = *(const uint4*)(lbrow + 16 + 8 * hi);
#pragma unroll
    for (int t = 0; t < 2; ++t) {
        Frag a;
        const _Float16* tp = tmp + (size_t)(mh + t * 16 + lr) * LORA_R + 8 * hi;
        a.q[0] = *(const uint4*)tp;
        a.q[1] = *(const uint4*)(tp + 16);
        acc[t] = wmma_f16(a, lb, acc[t]);
    }

    // ---- store f16 (streamed once -> non-temporal) ----
#pragma unroll
    for (int t = 0; t < 2; ++t)
#pragma unroll
        for (int v = 0; v < 8; ++v) {
            int m = mh + t * 16 + v + 8 * hi;      // C layout: lanes 16-31 hold M+8
            __builtin_nontemporal_store((_Float16)acc[t][v],
                                        out + (size_t)m * OUT_DIM + n);
        }
}

extern "C" void kernel_launch(void* const* d_in, const int* in_sizes, int n_in,
                              void* d_out, int out_size, void* d_ws, size_t ws_size,
                              hipStream_t stream) {
    const _Float16* x      = (const _Float16*)d_in[0];
    const int8_t*   w      = (const int8_t*)  d_in[1];
    const _Float16* wscale = (const _Float16*)d_in[2];
    const _Float16* lora_a = (const _Float16*)d_in[3];
    const _Float16* lora_b = (const _Float16*)d_in[4];
    _Float16* out = (_Float16*)d_out;
    _Float16* tmp = (_Float16*)d_ws;             // 64*32 f16 = 4 KB scratch

    lora_xa_kernel<<<1, 256, 0, stream>>>(x, lora_a, tmp);
    lora_qlinear_kernel<<<OUT_DIM / NBLK, 256, 0, stream>>>(x, w, wscale, lora_b, tmp, out);
}
